// GatedSTGCNBlock_52785148068461
// MI455X (gfx1250) — matrix operand order repaired and checked by reading, not compiled, not run on hardware
//
#include <hip/hip_runtime.h>
#include <math.h>

typedef __attribute__((ext_vector_type(16))) _Float16 v16h;
typedef __attribute__((ext_vector_type(8)))  _Float16 v8h;
typedef __attribute__((ext_vector_type(8)))  float    v8f;

#define T_DIM   64
#define N_NODES 2048
#define C_DIM   64
#define E_EDGES 32768
#define ROWS    (T_DIM * N_NODES)   // 131072

// workspace layout (in floats)
#define WS_XW    0
#define WS_DEG   (ROWS * C_DIM)              // 8388608
#define WS_DINV  (WS_DEG  + N_NODES)
#define WS_IDEG  (WS_DINV + N_NODES)
#define WS_NORM  (WS_IDEG + N_NODES)
#define WS_STATS (WS_NORM + E_EDGES)         // 128 floats: sum[64], sumsq[64]

// ---------------------------------------------------------------- utilities
__global__ __launch_bounds__(256) void k_init(float* deg, float* stats) {
    int i = blockIdx.x * 256 + threadIdx.x;
    if (i < N_NODES) deg[i] = 1.0f;          // self-loop weight
    if (i < 128)     stats[i] = 0.0f;
}

__global__ __launch_bounds__(256) void k_deg(const int* __restrict__ ei,
                                             const float* __restrict__ ew,
                                             float* deg) {
    int e = blockIdx.x * 256 + threadIdx.x;
    if (e < E_EDGES) atomicAdd(&deg[ei[E_EDGES + e]], ew[e]);
}

__global__ __launch_bounds__(256) void k_dinv(const float* __restrict__ deg,
                                              float* dinv, float* ideg) {
    int n = blockIdx.x * 256 + threadIdx.x;
    if (n < N_NODES) {
        float d = deg[n];
        dinv[n] = rsqrtf(d);
        ideg[n] = 1.0f / d;
    }
}

__global__ __launch_bounds__(256) void k_norm(const int* __restrict__ ei,
                                              const float* __restrict__ ew,
                                              const float* __restrict__ dinv,
                                              float* nrm) {
    int e = blockIdx.x * 256 + threadIdx.x;
    if (e < E_EDGES) nrm[e] = dinv[ei[e]] * ew[e] * dinv[ei[E_EDGES + e]];
}

// ------------------------------------------- fused gated-conv + GCN linear
// Each wave: 16 rows x 64 cols.  rows = t*2048 + n (tiles never cross t).
__global__ __launch_bounds__(128) void k_conv_gcn(
    const float* __restrict__ x,
    const float* __restrict__ conv_w, const float* __restrict__ conv_b,
    const float* __restrict__ gate_w, const float* __restrict__ gate_b,
    const float* __restrict__ gcn_w,  const float* __restrict__ gcn_b,
    const float* __restrict__ invdeg,
    float* __restrict__ xw_out, float* __restrict__ outpre)
{
    // B-matrix layout in LDS: w[n][k] so each lane reads 16 contiguous halves
    __shared__ __attribute__((aligned(32))) _Float16 wC[3][64][64];
    __shared__ __attribute__((aligned(32))) _Float16 wG[3][64][64];
    __shared__ __attribute__((aligned(32))) _Float16 wX[64][64];
    __shared__ __attribute__((aligned(32))) _Float16 hst[4][16][64];

    const int tid = threadIdx.x;
    for (int i = tid; i < 3 * 64 * 64; i += 128) {
        int co = i / 192, r = i % 192, ci = r / 3, tap = r % 3;
        wC[tap][co][ci] = (_Float16)conv_w[i];
        wG[tap][co][ci] = (_Float16)gate_w[i];
    }
    for (int i = tid; i < 64 * 64; i += 128) {
        int k = i >> 6, n = i & 63;            // gcn_w[k][n] -> B[k][n]
        wX[n][k] = (_Float16)gcn_w[i];
    }
    __syncthreads();

    const int wave  = tid >> 5, lane = tid & 31;
    const int rlane = lane & 15, grp = lane >> 4, kbase = grp * 8;
    const int rowbase = blockIdx.x * 64 + wave * 16;
    const int t = rowbase >> 11;               // N = 2048
    const int nbase = rowbase & 2047;

    v8f accC[4] = {}, accG[4] = {};

    #pragma unroll
    for (int tap = 0; tap < 3; ++tap) {
        int tt = t + 2 * tap - 2;              // taps at t-2, t, t+2
        if (tt < 0 || tt >= T_DIM) continue;   // wave-uniform branch
        const float* xr = x + (size_t)(tt * N_NODES + nbase + rlane) * C_DIM;
        #pragma unroll
        for (int s = 0; s < 2; ++s) {          // K = 64 -> 2 steps of 32
            // A layout: a[0..15]=K grp*16.. (contiguous, matches B)
            const float* p0 = xr + s * 32 + grp * 16;
            v16h a;
            #pragma unroll
            for (int i = 0; i < 8; ++i) {
                a[i]     = (_Float16)p0[i];
                a[8 + i] = (_Float16)p0[8 + i];
            }
            #pragma unroll
            for (int nt = 0; nt < 4; ++nt) {
                v16h bC = *(const v16h*)&wC[tap][nt * 16 + rlane][s * 32 + grp * 16];
                v16h bG = *(const v16h*)&wG[tap][nt * 16 + rlane][s * 32 + grp * 16];
                accC[nt] = __builtin_amdgcn_wmma_f32_16x16x32_f16(
                    false, a, false, bC, (short)0, accC[nt], false, false);
                accG[nt] = __builtin_amdgcn_wmma_f32_16x16x32_f16(
                    false, a, false, bG, (short)0, accG[nt], false, false);
            }
        }
    }

    // gated activation, restage h into A-friendly LDS tile (per-wave region)
    #pragma unroll
    for (int nt = 0; nt < 4; ++nt) {
        int col = nt * 16 + rlane;
        float cb = conv_b[col], gb = gate_b[col];
        #pragma unroll
        for (int v = 0; v < 8; ++v) {
            int m = grp * 8 + v;               // C layout: M = 8*grp + v
            float cv = accC[nt][v] + cb;
            float gv = accG[nt][v] + gb;
            float hv = cv * (1.0f / (1.0f + __expf(-gv)));
            hst[wave][m][col] = (_Float16)hv;
        }
    }

    // second GEMM: xw = h @ gcn_w
    v8f accX[4] = {};
    #pragma unroll
    for (int s = 0; s < 2; ++s) {
        const _Float16* hr = &hst[wave][rlane][0];
        v8h lo = *(const v8h*)(hr + s * 32 + grp * 16);
        v8h hi = *(const v8h*)(hr + s * 32 + grp * 16 + 8);
        v16h a;
        #pragma unroll
        for (int i = 0; i < 8; ++i) { a[i] = lo[i]; a[8 + i] = hi[i]; }
        #pragma unroll
        for (int nt = 0; nt < 4; ++nt) {
            v16h bX = *(const v16h*)&wX[nt * 16 + rlane][s * 32 + grp * 16];
            accX[nt] = __builtin_amdgcn_wmma_f32_16x16x32_f16(
                false, a, false, bX, (short)0, accX[nt], false, false);
        }
    }

    // epilogue: store xw, init output with self-loop term xw/deg + bias
    float idg[8];
    #pragma unroll
    for (int v = 0; v < 8; ++v) idg[v] = invdeg[nbase + grp * 8 + v];
    #pragma unroll
    for (int nt = 0; nt < 4; ++nt) {
        int col = nt * 16 + rlane;
        float gb = gcn_b[col];
        #pragma unroll
        for (int v = 0; v < 8; ++v) {
            size_t row = (size_t)(rowbase + grp * 8 + v);
            float xv = accX[nt][v];
            xw_out[row * 64 + col] = xv;
            outpre[row * 64 + col] = xv * idg[v] + gb;
        }
    }
}

// ---------------------------------------------------------- edge scatter
__global__ __launch_bounds__(64) void k_scatter(
    const float* __restrict__ xw, const int* __restrict__ ei,
    const float* __restrict__ nrm, float* __restrict__ out)
{
    int e = blockIdx.x, t = blockIdx.y, c = threadIdx.x;
    int src = ei[e], dst = ei[E_EDGES + e];
    float nm = nrm[e];
    float v = xw[(size_t)(t * N_NODES + src) * 64 + c] * nm;
    atomicAdd(&out[(size_t)(t * N_NODES + dst) * 64 + c], v);
}

// ------------------------------------------------------------- BN stats
__global__ __launch_bounds__(256) void k_stats(const float* __restrict__ out,
                                               float* __restrict__ stats)
{
    __shared__ float s1[256], s2[256];
    int tid = threadIdx.x, c = tid & 63, rs = tid >> 6;
    float a = 0.0f, b = 0.0f;
    for (int row = blockIdx.x * 4 + rs; row < ROWS; row += gridDim.x * 4) {
        float v = out[(size_t)row * 64 + c];
        a += v; b += v * v;
    }
    s1[tid] = a; s2[tid] = b;
    __syncthreads();
    if (rs == 0) {
        a = s1[c] + s1[64 + c] + s1[128 + c] + s1[192 + c];
        b = s2[c] + s2[64 + c] + s2[128 + c] + s2[192 + c];
        atomicAdd(&stats[c], a);
        atomicAdd(&stats[64 + c], b);
    }
}

__global__ __launch_bounds__(256) void k_bn_relu(
    float* __restrict__ out, const float* __restrict__ stats,
    const float* __restrict__ gamma, const float* __restrict__ beta)
{
    size_t idx = (size_t)blockIdx.x * 256 + threadIdx.x;
    int c = (int)(idx & 63);
    const float cnt = (float)ROWS;
    float mean = stats[c] / cnt;
    float var  = stats[64 + c] / cnt - mean * mean;
    float v = (out[idx] - mean) * rsqrtf(var + 1e-5f) * gamma[c] + beta[c];
    out[idx] = fmaxf(v, 0.0f);
}

// -------------------------------------------------------------- launcher
extern "C" void kernel_launch(void* const* d_in, const int* in_sizes, int n_in,
                              void* d_out, int out_size, void* d_ws, size_t ws_size,
                              hipStream_t stream)
{
    const float* x      = (const float*)d_in[0];
    const int*   ei     = (const int*)  d_in[1];
    const float* ew     = (const float*)d_in[2];
    const float* conv_w = (const float*)d_in[3];
    const float* conv_b = (const float*)d_in[4];
    const float* gate_w = (const float*)d_in[5];
    const float* gate_b = (const float*)d_in[6];
    const float* gcn_w  = (const float*)d_in[7];
    const float* gcn_b  = (const float*)d_in[8];
    const float* gamma  = (const float*)d_in[9];
    const float* beta   = (const float*)d_in[10];

    float* out = (float*)d_out;
    float* ws  = (float*)d_ws;

    float* xw    = ws + WS_XW;
    float* deg   = ws + WS_DEG;
    float* dinv  = ws + WS_DINV;
    float* ideg  = ws + WS_IDEG;
    float* nrm   = ws + WS_NORM;
    float* stats = ws + WS_STATS;

    k_init<<<(N_NODES + 255) / 256, 256, 0, stream>>>(deg, stats);
    k_deg <<<(E_EDGES + 255) / 256, 256, 0, stream>>>(ei, ew, deg);
    k_dinv<<<(N_NODES + 255) / 256, 256, 0, stream>>>(deg, dinv, ideg);
    k_norm<<<(E_EDGES + 255) / 256, 256, 0, stream>>>(ei, ew, dinv, nrm);

    k_conv_gcn<<<ROWS / 64, 128, 0, stream>>>(x, conv_w, conv_b, gate_w, gate_b,
                                              gcn_w, gcn_b, ideg, xw, out);

    dim3 sg(E_EDGES, T_DIM);
    k_scatter<<<sg, 64, 0, stream>>>(xw, ei, nrm, out);

    k_stats<<<256, 256, 0, stream>>>(out, stats);
    k_bn_relu<<<(ROWS * C_DIM) / 256, 256, 0, stream>>>(out, stats, gamma, beta);
}
